// HyTE_57037165691116
// MI455X (gfx1250) — matrix operand order repaired and checked by reading, not compiled, not run on hardware
//
#include <hip/hip_runtime.h>
#include <hip/hip_bf16.h>

#define N_NODES 100000
#define ENT     50000
#define DIM     128
#define N_EDGES 1600000
#define BATCH   131072
#define REL     500
#define TIMEN   365
#define EPS_F   1e-12f

typedef __attribute__((ext_vector_type(16))) __bf16 v16bf;
typedef __attribute__((ext_vector_type(8)))  float  v8f;

// ---------------------------------------------------------------- utilities
__device__ __forceinline__ float wave_reduce_sum(float v) {
#pragma unroll
    for (int off = 16; off > 0; off >>= 1)
        v += __shfl_xor(v, off, 32);
    return v;
}

// ---------------------------------------------------------------- zero fill
__global__ void zero_f32(float* __restrict__ p, int n) {
    int i = blockIdx.x * blockDim.x + threadIdx.x;
    if (i < n) p[i] = 0.0f;
}

// ------------------------------------------------- W -> bf16 split, WMMA order
// B[k][n] = W[n][k]  (out = agg @ W^T).  Stored in register order:
// index = (((kt*8 + nt)*32 + lane)*16 + j), lane: n = (lane&15)+16*nt,
// K = 32*kt + 16*(lane>>4) + j   (j = 2v+odd matches VGPR v pack {K,K+1}).
__global__ void prep_w(const float* __restrict__ W,
                       __bf16* __restrict__ Bh, __bf16* __restrict__ Bl) {
    int kt   = blockIdx.x;          // 0..3
    int nt   = threadIdx.x >> 5;    // 0..7
    int lane = threadIdx.x & 31;
    int n    = (lane & 15) + 16 * nt;
    int kh   = 32 * kt + 16 * (lane >> 4);
    int base = (((kt * 8 + nt) * 32) + lane) * 16;
#pragma unroll
    for (int j = 0; j < 16; ++j) {
        float w    = W[n * DIM + kh + j];
        __bf16 hi  = (__bf16)w;
        Bh[base + j] = hi;
        Bl[base + j] = (__bf16)(w - (float)hi);
    }
}

// ------------------------------------------------- edge gather + scatter-add
// one wave per edge; 32 lanes x float4 = 128 dims; atomics land in L2
__global__ void scatter_edges(const float* __restrict__ feat,
                              const int* __restrict__ src,
                              const int* __restrict__ dst,
                              float* __restrict__ agg) {
    int wave = blockIdx.x * (blockDim.x >> 5) + (threadIdx.x >> 5);
    if (wave >= N_EDGES) return;
    int lane = threadIdx.x & 31;
    int s = src[wave], d = dst[wave];
    const float4 v = *(const float4*)(feat + (size_t)s * DIM + lane * 4);
    float* o = agg + (size_t)d * DIM + lane * 4;
    atomicAdd(o + 0, v.x);
    atomicAdd(o + 1, v.y);
    atomicAdd(o + 2, v.z);
    atomicAdd(o + 3, v.w);
}

// ------------------------------------------------- WMMA GEMM + bias + ReLU
// out[16 rows x 128 cols] per wave; K=128 via 4x v_wmma_f32_16x16x32_bf16,
// 3-term bf16 split (Ah*Bh + Ah*Bl + Al*Bh) ~ f32 accuracy.
__global__ void __launch_bounds__(128)
gcn_gemm_relu(const float* __restrict__ A,
              const __bf16* __restrict__ Bh, const __bf16* __restrict__ Bl,
              const float* __restrict__ bias, float* __restrict__ out) {
    __shared__ alignas(32) __bf16 sBh[16384];   // 32 KB
    __shared__ alignas(32) __bf16 sBl[16384];   // 32 KB

    // cooperative stage of the split weights into LDS (8192 dwords each)
    {
        const uint4* gh = (const uint4*)Bh;
        const uint4* gl = (const uint4*)Bl;
        uint4* sh = (uint4*)sBh;
        uint4* sl = (uint4*)sBl;
        for (int i = threadIdx.x; i < 2048; i += 128) {
            sh[i] = gh[i];
            sl[i] = gl[i];
        }
    }
    __syncthreads();

    int wave = blockIdx.x * 4 + (threadIdx.x >> 5);
    if (wave >= N_NODES / 16) return;           // wave-uniform: EXEC stays full
    int lane = threadIdx.x & 31;
    int half = lane >> 4;
    int m    = wave * 16 + (lane & 15);
    const float* arow = A + (size_t)m * DIM;

    v8f acc[8];
    v8f zero = {};
#pragma unroll
    for (int i = 0; i < 8; ++i) acc[i] = zero;

#pragma unroll
    for (int kt = 0; kt < 4; ++kt) {
        // A tile: 16x32 bf16, K = 32*kt + 16*half + j (matches B), split into hi/lo
        v16bf ah, al;
#pragma unroll
        for (int v = 0; v < 8; ++v) {
            int k = 32 * kt + 16 * half + 2 * v;
            float2 f = *(const float2*)(arow + k);
            __bf16 h0 = (__bf16)f.x;
            __bf16 h1 = (__bf16)f.y;
            ah[2 * v]     = h0;
            ah[2 * v + 1] = h1;
            al[2 * v]     = (__bf16)(f.x - (float)h0);
            al[2 * v + 1] = (__bf16)(f.y - (float)h1);
        }
#pragma unroll
        for (int nt = 0; nt < 8; ++nt) {
            int tb = (((kt * 8 + nt) * 32) + lane) * 16;
            v16bf bh = *(const v16bf*)(sBh + tb);
            v16bf bl = *(const v16bf*)(sBl + tb);
            acc[nt] = __builtin_amdgcn_wmma_f32_16x16x32_bf16(
                false, ah, false, bh, (short)0, acc[nt], false, false);
            acc[nt] = __builtin_amdgcn_wmma_f32_16x16x32_bf16(
                false, ah, false, bl, (short)0, acc[nt], false, false);
            acc[nt] = __builtin_amdgcn_wmma_f32_16x16x32_bf16(
                false, al, false, bh, (short)0, acc[nt], false, false);
        }
    }

    // epilogue: bias + ReLU; C/D layout: row = v + 8*half, col = lane&15
#pragma unroll
    for (int nt = 0; nt < 8; ++nt) {
        int n = (lane & 15) + 16 * nt;
        float bb = bias[n];
#pragma unroll
        for (int v = 0; v < 8; ++v) {
            int row = wave * 16 + v + 8 * half;
            out[(size_t)row * DIM + n] = fmaxf(acc[nt][v] + bb, 0.0f);
        }
    }
}

// ------------------------------------------------- snapshot mean (SNAP=2)
__global__ void ent_mean(const float* __restrict__ f, float* __restrict__ ent) {
    int i = blockIdx.x * blockDim.x + threadIdx.x;
    if (i < ENT * DIM) ent[i] = 0.5f * (f[i] + f[i + ENT * DIM]);
}

// ------------------------------------------------- batch scoring
__device__ __forceinline__ float4 proj_norm(float4 e, float4 n) {
    float d = wave_reduce_sum(e.x * n.x + e.y * n.y + e.z * n.z + e.w * n.w);
    e.x -= d * n.x; e.y -= d * n.y; e.z -= d * n.z; e.w -= d * n.w;
    float en = sqrtf(wave_reduce_sum(e.x * e.x + e.y * e.y + e.z * e.z + e.w * e.w));
    float inv = 1.0f / fmaxf(en, EPS_F);
    e.x *= inv; e.y *= inv; e.z *= inv; e.w *= inv;
    return e;
}

__global__ void hyte_score(const float* __restrict__ ent,
                           const float* __restrict__ rel,
                           const float* __restrict__ nrm,
                           const int* __restrict__ h_idx,
                           const int* __restrict__ r_idx,
                           const int* __restrict__ t_idx,
                           const int* __restrict__ tm_idx,
                           float* __restrict__ out) {
    int row = blockIdx.x * (blockDim.x >> 5) + (threadIdx.x >> 5);
    if (row >= BATCH) return;
    int lane = threadIdx.x & 31;
    int c = lane * 4;

    float4 h = *(const float4*)(ent + (size_t)h_idx[row] * DIM + c);
    float4 r = *(const float4*)(rel + (size_t)r_idx[row] * DIM + c);
    float4 t = *(const float4*)(ent + (size_t)t_idx[row] * DIM + c);
    float4 n = *(const float4*)(nrm + (size_t)tm_idx[row] * DIM + c);

    float nn = sqrtf(wave_reduce_sum(n.x * n.x + n.y * n.y + n.z * n.z + n.w * n.w));
    float inv = 1.0f / fmaxf(nn, EPS_F);
    n.x *= inv; n.y *= inv; n.z *= inv; n.w *= inv;

    h = proj_norm(h, n);
    r = proj_norm(r, n);
    t = proj_norm(t, n);

    float sx = h.x + r.x - t.x;
    float sy = h.y + r.y - t.y;
    float sz = h.z + r.z - t.z;
    float sw = h.w + r.w - t.w;
    float sc = sqrtf(wave_reduce_sum(sx * sx + sy * sy + sz * sz + sw * sw));
    if (lane == 0) out[row] = sc;
}

// ---------------------------------------------------------------- launcher
extern "C" void kernel_launch(void* const* d_in, const int* in_sizes, int n_in,
                              void* d_out, int out_size, void* d_ws, size_t ws_size,
                              hipStream_t stream) {
    (void)in_sizes; (void)n_in; (void)out_size; (void)ws_size;

    const float* node_feat = (const float*)d_in[0];
    const float* W         = (const float*)d_in[1];
    const float* bvec      = (const float*)d_in[2];
    const float* rel_emb   = (const float*)d_in[3];
    const float* norm_emb  = (const float*)d_in[4];
    const int*   src       = (const int*)d_in[5];
    const int*   dst       = (const int*)d_in[6];
    const int*   h_idx     = (const int*)d_in[7];
    const int*   r_idx     = (const int*)d_in[8];
    const int*   t_idx     = (const int*)d_in[9];
    const int*   tm_idx    = (const int*)d_in[10];
    float*       out       = (float*)d_out;

    char* ws = (char*)d_ws;
    size_t off = 0;
    float*  agg = (float*)(ws + off);  off += (size_t)N_NODES * DIM * sizeof(float); // 51.2 MB
    float*  h1  = (float*)(ws + off);  off += (size_t)N_NODES * DIM * sizeof(float); // 51.2 MB
    float*  ent = (float*)(ws + off);  off += (size_t)ENT * DIM * sizeof(float);     // 25.6 MB
    __bf16* Bh  = (__bf16*)(ws + off); off += 16384 * sizeof(__bf16);                // 32 KB
    __bf16* Bl  = (__bf16*)(ws + off);                                               // 32 KB

    const int nfeat = N_NODES * DIM;
    const int gemm_blocks = (N_NODES / 16 + 3) / 4;   // 4 waves/block

    prep_w<<<4, 256, 0, stream>>>(W, Bh, Bl);

    // layer 1
    zero_f32<<<(nfeat + 255) / 256, 256, 0, stream>>>(agg, nfeat);
    scatter_edges<<<(N_EDGES + 7) / 8, 256, 0, stream>>>(node_feat, src, dst, agg);
    gcn_gemm_relu<<<gemm_blocks, 128, 0, stream>>>(agg, Bh, Bl, bvec, h1);

    // layer 2 (shared weights); output overwrites h1 (agg is the only read src)
    zero_f32<<<(nfeat + 255) / 256, 256, 0, stream>>>(agg, nfeat);
    scatter_edges<<<(N_EDGES + 7) / 8, 256, 0, stream>>>(h1, src, dst, agg);
    gcn_gemm_relu<<<gemm_blocks, 128, 0, stream>>>(agg, Bh, Bl, bvec, h1);

    // temporal mean over SNAP=2, then batched hyperplane scoring
    ent_mean<<<(ENT * DIM + 255) / 256, 256, 0, stream>>>(h1, ent);
    hyte_score<<<(BATCH + 7) / 8, 256, 0, stream>>>(ent, rel_emb, norm_emb,
                                                    h_idx, r_idx, t_idx, tm_idx, out);
}
